// CrossWindowAttention3D_15796889714999
// MI455X (gfx1250) — matrix operand order repaired and checked by reading, not compiled, not run on hardware
//
#include <hip/hip_runtime.h>

typedef __attribute__((ext_vector_type(16))) _Float16 v16h;
typedef __attribute__((ext_vector_type(8)))  float    v8f;

#define NH      6
#define NTOK    98
#define NPAD    128
#define CDIM    192
#define NB      1024
#define NWIN    64
#define HDIM    32

// ---------------------------------------------------------------------------
// Prep: convert 4 weight matrices f32 -> f16 (packed back-to-back in ws)
// ---------------------------------------------------------------------------
__global__ void cvt_w_kernel(const float* __restrict__ wq, const float* __restrict__ wk,
                             const float* __restrict__ wv, const float* __restrict__ wp,
                             _Float16* __restrict__ dst) {
    int i = blockIdx.x * blockDim.x + threadIdx.x;
    if (i < CDIM * CDIM) {
        dst[i]                   = (_Float16)wq[i];
        dst[CDIM * CDIM + i]     = (_Float16)wk[i];
        dst[2 * CDIM * CDIM + i] = (_Float16)wv[i];
        dst[3 * CDIM * CDIM + i] = (_Float16)wp[i];
    }
}

// ---------------------------------------------------------------------------
// Prep: gather relative-position bias  ->  Bias[h][98][98]
// ---------------------------------------------------------------------------
__global__ void bias_kernel(const float* __restrict__ table, const int* __restrict__ rel,
                            float* __restrict__ bias) {
    int idx = blockIdx.x * blockDim.x + threadIdx.x;   // n*98 + m
    if (idx < NTOK * NTOK) {
        int r = rel[idx];
#pragma unroll
        for (int h = 0; h < NH; ++h)
            bias[h * NTOK * NTOK + idx] = table[r * NH + h];
    }
}

// ---------------------------------------------------------------------------
// Projection: dst = (src @ W^T + b) * scale, split into heads.
// One block per window, 7 waves = 7 row tiles (rows 0..111, stores guarded <98).
// vmode 0: dst[(b*6+h)*128 + n][d]   (Q, K natural)
// vmode 1: dst[(b*6+h)*32  + d][n]   (V transposed)
// ---------------------------------------------------------------------------
__global__ __launch_bounds__(224)
void proj_kernel(const float* __restrict__ src, const _Float16* __restrict__ W,
                 const float* __restrict__ bias, float scale,
                 _Float16* __restrict__ dst, int vmode) {
    const int b    = blockIdx.x;
    const int wid  = threadIdx.x >> 5;
    const int lane = threadIdx.x & 31;
    const int nn   = lane & 15;
    const int g    = lane >> 4;
    const int rt   = wid;

    const int arow = rt * 16 + nn;                 // A-fragment row (all lanes: M = lane%16)
    const bool rvalid = arow < NTOK;
    const float* srow = src + ((size_t)b * NTOK + (rvalid ? arow : 0)) * CDIM;

    v8f acc[12];
#pragma unroll
    for (int ct = 0; ct < 12; ++ct)
#pragma unroll
        for (int i = 0; i < 8; ++i) acc[ct][i] = 0.f;

    for (int k0 = 0; k0 < CDIM; k0 += 32) {
        v16h a;
        if (rvalid) {
            const float* p0 = srow + k0 + 16 * g;       // K = k0+16g .. +7
            const float* p1 = srow + k0 + 16 * g + 8;   // K = k0+16g+8 .. +7
#pragma unroll
            for (int i = 0; i < 8; ++i) { a[i] = (_Float16)p0[i]; a[8 + i] = (_Float16)p1[i]; }
        } else {
#pragma unroll
            for (int i = 0; i < 16; ++i) a[i] = (_Float16)0.f;
        }
#pragma unroll
        for (int ct = 0; ct < 12; ++ct) {
            // B[k][n] = W[n][k]  -> lane reads 16 contiguous halfs of W row (32B aligned)
            v16h bf = *(const v16h*)(W + (size_t)(ct * 16 + nn) * CDIM + k0 + 16 * g);
            acc[ct] = __builtin_amdgcn_wmma_f32_16x16x32_f16(false, a, false, bf,
                                                             (short)0, acc[ct], false, false);
        }
    }

#pragma unroll
    for (int ct = 0; ct < 12; ++ct) {
        const int c  = ct * 16 + nn;
        const int h  = c >> 5;
        const int d  = c & 31;
        const float bv = bias[c];
        const size_t bh = (size_t)b * NH + h;
#pragma unroll
        for (int i = 0; i < 8; ++i) {
            const int orow = rt * 16 + 8 * g + i;       // D row = i + 8*(lane/16)
            if (orow < NTOK) {
                const float v = (acc[ct][i] + bv) * scale;
                if (vmode == 0) dst[(bh * NPAD + orow) * HDIM + d] = (_Float16)v;
                else            dst[(bh * HDIM + d) * NPAD + orow] = (_Float16)v;
            }
        }
    }
}

// ---------------------------------------------------------------------------
// Attention: one block per (b,h). 7 waves = 7 query row tiles.
// S = Q K^T (1 WMMA / 16x16 tile, K-dim=32) ; +bias +mask ; softmax ;
// P -> LDS ; O = P V (2 col tiles x 4 k-steps, K padded to 128).
// ---------------------------------------------------------------------------
__global__ __launch_bounds__(224)
void attn_kernel(const _Float16* __restrict__ Qh, const _Float16* __restrict__ Kh,
                 const _Float16* __restrict__ Vt, const float* __restrict__ Bias,
                 const float* __restrict__ Mask, _Float16* __restrict__ O) {
    __shared__ _Float16 Pls[7 * 16 * 112];

    const int bh   = blockIdx.x;
    const int b    = bh / NH;
    const int h    = bh % NH;
    const int w    = b & (NWIN - 1);
    const int wid  = threadIdx.x >> 5;
    const int lane = threadIdx.x & 31;
    const int nn   = lane & 15;
    const int g    = lane >> 4;
    const int rt   = wid;

    const _Float16* Qb = Qh + (size_t)bh * NPAD * HDIM;
    const _Float16* Kb = Kh + (size_t)bh * NPAD * HDIM;
    const _Float16* Vb = Vt + (size_t)bh * HDIM * NPAD;
    const float*    Bh = Bias + (size_t)h * NTOK * NTOK;
    const float*    Mw = Mask + (size_t)w * NTOK * NTOK;

    // A fragment: Q rows rt*16 + (lane%16), two contiguous 8-half runs
    v16h a;
    {
        const _Float16* q0 = Qb + (size_t)(rt * 16 + nn) * HDIM + 16 * g;
#pragma unroll
        for (int i = 0; i < 8; ++i) { a[i] = q0[i]; a[8 + i] = q0[8 + i]; }
    }

    // S = Q K^T : 7 tiles, single 16x16x32 WMMA each
    v8f s[7];
#pragma unroll
    for (int ct = 0; ct < 7; ++ct) {
        v16h bf = *(const v16h*)(Kb + (size_t)(ct * 16 + nn) * HDIM + 16 * g);
        v8f z;
#pragma unroll
        for (int i = 0; i < 8; ++i) z[i] = 0.f;
        s[ct] = __builtin_amdgcn_wmma_f32_16x16x32_f16(false, a, false, bf,
                                                       (short)0, z, false, false);
    }

    // bias + mask (valid cols only)
#pragma unroll
    for (int ct = 0; ct < 7; ++ct) {
        const int col = ct * 16 + nn;
        const bool cvalid = col < NTOK;
#pragma unroll
        for (int i = 0; i < 8; ++i) {
            int row = rt * 16 + 8 * g + i;
            int rc  = row < NTOK ? row : NTOK - 1;     // clamp (rows>=98 unused)
            if (cvalid) s[ct][i] += Bh[rc * NTOK + col] + Mw[rc * NTOK + col];
        }
    }

    // softmax over columns (row lives across the 16 lanes of this half-wave group)
    const bool vlast = (nn < 2);                       // ct==6 -> cols 96..111, valid iff nn<2
    _Float16* Pt = &Pls[wid * 16 * 112];
#pragma unroll
    for (int i = 0; i < 8; ++i) {
        float m = -3.0e38f;
#pragma unroll
        for (int ct = 0; ct < 6; ++ct) m = fmaxf(m, s[ct][i]);
        if (vlast) m = fmaxf(m, s[6][i]);
#pragma unroll
        for (int off = 1; off < 16; off <<= 1) m = fmaxf(m, __shfl_xor(m, off, 32));
        float e[7], sum = 0.f;
#pragma unroll
        for (int ct = 0; ct < 6; ++ct) { e[ct] = __expf(s[ct][i] - m); sum += e[ct]; }
        e[6] = vlast ? __expf(s[6][i] - m) : 0.f;
        sum += e[6];
#pragma unroll
        for (int off = 1; off < 16; off <<= 1) sum += __shfl_xor(sum, off, 32);
        const float inv = __builtin_amdgcn_rcpf(sum);
        const int prow = 8 * g + i;
#pragma unroll
        for (int ct = 0; ct < 7; ++ct)
            Pt[prow * 112 + ct * 16 + nn] = (_Float16)(e[ct] * inv);
    }
    __syncthreads();

    // O = P V : K = 128 (cols 98..111 are zero from softmax, 112..127 zeroed here)
#pragma unroll
    for (int co = 0; co < 2; ++co) {
        v8f o;
#pragma unroll
        for (int i = 0; i < 8; ++i) o[i] = 0.f;
#pragma unroll
        for (int ks = 0; ks < 4; ++ks) {
            const int k0 = ks * 32;
            v16h pa;
            const _Float16* pr = Pt + nn * 112 + k0 + 8 * g;
            if (ks < 3) {
#pragma unroll
                for (int i = 0; i < 8; ++i) { pa[i] = pr[i]; pa[8 + i] = pr[16 + i]; }
            } else {
#pragma unroll
                for (int i = 0; i < 8; ++i) { pa[i] = pr[i]; pa[8 + i] = (_Float16)0.f; }
            }
            v16h vb;
            const _Float16* vr = Vb + (size_t)(co * 16 + nn) * NPAD + k0 + 8 * g;
#pragma unroll
            for (int i = 0; i < 8; ++i) { vb[i] = vr[i]; vb[8 + i] = vr[16 + i]; }
            o = __builtin_amdgcn_wmma_f32_16x16x32_f16(false, pa, false, vb,
                                                       (short)0, o, false, false);
        }
#pragma unroll
        for (int i = 0; i < 8; ++i) {
            const int row = rt * 16 + 8 * g + i;
            if (row < NTOK)
                O[((size_t)b * NTOK + row) * CDIM + h * HDIM + co * 16 + nn] = (_Float16)o[i];
        }
    }
}

// ---------------------------------------------------------------------------
// Output projection: out = O @ Wp^T + bp  (f16 in, f32 out)
// ---------------------------------------------------------------------------
__global__ __launch_bounds__(224)
void outproj_kernel(const _Float16* __restrict__ O16, const _Float16* __restrict__ W,
                    const float* __restrict__ bp, float* __restrict__ out) {
    const int b    = blockIdx.x;
    const int wid  = threadIdx.x >> 5;
    const int lane = threadIdx.x & 31;
    const int nn   = lane & 15;
    const int g    = lane >> 4;
    const int rt   = wid;

    const int arow = rt * 16 + nn;
    const bool rvalid = arow < NTOK;
    const _Float16* srow = O16 + ((size_t)b * NTOK + (rvalid ? arow : 0)) * CDIM;

    v8f acc[12];
#pragma unroll
    for (int ct = 0; ct < 12; ++ct)
#pragma unroll
        for (int i = 0; i < 8; ++i) acc[ct][i] = 0.f;

    for (int k0 = 0; k0 < CDIM; k0 += 32) {
        v16h a;
        if (rvalid) {
            const _Float16* p0 = srow + k0 + 16 * g;
            const _Float16* p1 = srow + k0 + 16 * g + 8;
#pragma unroll
            for (int i = 0; i < 8; ++i) { a[i] = p0[i]; a[8 + i] = p1[i]; }
        } else {
#pragma unroll
            for (int i = 0; i < 16; ++i) a[i] = (_Float16)0.f;
        }
#pragma unroll
        for (int ct = 0; ct < 12; ++ct) {
            v16h bf = *(const v16h*)(W + (size_t)(ct * 16 + nn) * CDIM + k0 + 16 * g);
            acc[ct] = __builtin_amdgcn_wmma_f32_16x16x32_f16(false, a, false, bf,
                                                             (short)0, acc[ct], false, false);
        }
    }

#pragma unroll
    for (int ct = 0; ct < 12; ++ct) {
        const int c = ct * 16 + nn;
        const float bv = bp[c];
#pragma unroll
        for (int i = 0; i < 8; ++i) {
            const int orow = rt * 16 + 8 * g + i;
            if (orow < NTOK)
                out[((size_t)b * NTOK + orow) * CDIM + c] = acc[ct][i] + bv;
        }
    }
}

// ---------------------------------------------------------------------------
extern "C" void kernel_launch(void* const* d_in, const int* in_sizes, int n_in,
                              void* d_out, int out_size, void* d_ws, size_t ws_size,
                              hipStream_t stream) {
    (void)in_sizes; (void)n_in; (void)out_size; (void)ws_size;
    const float* x     = (const float*)d_in[0];
    const float* y     = (const float*)d_in[1];
    const float* mask  = (const float*)d_in[2];
    const float* Wq    = (const float*)d_in[3];
    const float* bq    = (const float*)d_in[4];
    const float* Wk    = (const float*)d_in[5];
    const float* bk    = (const float*)d_in[6];
    const float* Wv    = (const float*)d_in[7];
    const float* bv    = (const float*)d_in[8];
    const float* Wp    = (const float*)d_in[9];
    const float* bp    = (const float*)d_in[10];
    const float* table = (const float*)d_in[11];
    const int*   rel   = (const int*)d_in[12];
    float* out = (float*)d_out;
    char*  ws  = (char*)d_ws;

    const size_t WSZ   = (size_t)CDIM * CDIM;                 // elements per weight
    _Float16* W16  = (_Float16*)ws;                           // 4 * 73728 B
    float*    Bias = (float*)(ws + 4 * WSZ * sizeof(_Float16));
    size_t off = 4 * WSZ * sizeof(_Float16) + (size_t)NH * NTOK * NTOK * sizeof(float);
    off = (off + 255) & ~(size_t)255;
    const size_t QKV_BYTES = (size_t)NB * NH * NPAD * HDIM * sizeof(_Float16); // 50.3 MB
    _Float16* Qh  = (_Float16*)(ws + off); off += QKV_BYTES;
    _Float16* Kh  = (_Float16*)(ws + off); off += QKV_BYTES;
    _Float16* Vt  = (_Float16*)(ws + off); off += QKV_BYTES;
    _Float16* O16 = (_Float16*)(ws + off);

    cvt_w_kernel<<<(CDIM * CDIM + 255) / 256, 256, 0, stream>>>(Wq, Wk, Wv, Wp, W16);
    bias_kernel<<<(NTOK * NTOK + 255) / 256, 256, 0, stream>>>(table, rel, Bias);

    const float scale = 0.17677669529663687f;  // 32^-0.5
    proj_kernel<<<NB, 224, 0, stream>>>(x, W16,            bq, scale, Qh, 0);
    proj_kernel<<<NB, 224, 0, stream>>>(y, W16 + WSZ,      bk, 1.0f,  Kh, 0);
    proj_kernel<<<NB, 224, 0, stream>>>(y, W16 + 2 * WSZ,  bv, 1.0f,  Vt, 1);

    attn_kernel<<<NB * NH, 224, 0, stream>>>(Qh, Kh, Vt, Bias, mask, O16);

    outproj_kernel<<<NB, 224, 0, stream>>>(O16, W16 + 3 * WSZ, bp, out);
}
